// TextGraph_32049045963096
// MI455X (gfx1250) — compile-verified, matrix operand order checked by reading, not run on hardware
//
#include <hip/hip_runtime.h>
#include <math.h>

// ---------------------------------------------------------------------------
// Problem constants: B=256, S=77, V=49408, D=1024;  M = B*S = 19712 rows.
// ---------------------------------------------------------------------------
#define TG_M   19712
#define TG_D   1024
#define TG_S   77
#define TG_B   256

typedef __attribute__((ext_vector_type(16))) __bf16 v16bf;
typedef __attribute__((ext_vector_type(8)))  __bf16 v8bf;
typedef __attribute__((ext_vector_type(8)))  float  v8f;

union BfFrag { v16bf v; v8bf h[2]; };

__device__ __forceinline__ unsigned short f2bf(float f) {
    unsigned int u = __float_as_uint(f);
    u += 0x7FFFu + ((u >> 16) & 1u);          // round-to-nearest-even
    return (unsigned short)(u >> 16);
}

// CDNA5 async global->LDS copy (VGLOBAL GLOBAL_LOAD_ASYNC_TO_LDS_B128, ASYNCcnt)
__device__ __forceinline__ void tg_async16(unsigned lds_off, const void* gsrc) {
    asm volatile("global_load_async_to_lds_b128 %0, %1, off"
                 :: "v"(lds_off), "v"(gsrc) : "memory");
}
__device__ __forceinline__ void tg_wait_async0() {
    asm volatile("s_wait_asynccnt 0x0" ::: "memory");
}

#define EPSF  1e-15f
#define MAXNF (1.0f - 1e-5f)

// ---------------------------------------------------------------------------
// 1) Convert + transpose weights to bf16, N-major: wt[mat][n][k] = w[mat][k][n]
//    mats 0..4 = mlp_w[0..4], mat 5 = gcn_w[3].
// ---------------------------------------------------------------------------
__global__ __launch_bounds__(256) void tg_convert_w(const float* __restrict__ mlp_w,
                                                    const float* __restrict__ gcn_w,
                                                    unsigned short* __restrict__ wt) {
    int idx = blockIdx.x * 256 + threadIdx.x;          // 6*1024*1024 exact
    int mat = idx >> 20;
    int rem = idx & 0xFFFFF;
    int n = rem >> 10;
    int k = rem & 1023;
    const float* src = (mat < 5) ? (mlp_w + (size_t)mat * 1048576)
                                 : (gcn_w + (size_t)3 * 1048576);
    wt[idx] = f2bf(src[k * 1024 + n]);
}

// ---------------------------------------------------------------------------
// 2) Embedding: x = tok_w[tokens[b,s]] + pos_w[s]  -> bf16
// ---------------------------------------------------------------------------
__global__ __launch_bounds__(256) void tg_embed(const long long* __restrict__ tokens,
                                                const float* __restrict__ tok_w,
                                                const float* __restrict__ pos_w,
                                                unsigned short* __restrict__ out) {
    int idx = blockIdx.x * 256 + threadIdx.x;          // 19712*1024 exact
    int row = idx >> 10;
    int d = idx & 1023;
    int s = row % TG_S;
    long long t = tokens[row];
    float v = tok_w[(size_t)t * TG_D + d] + pos_w[s * TG_D + d];
    out[idx] = f2bf(v);
}

// ---------------------------------------------------------------------------
// 3) bf16 WMMA GEMM:  C[M,1024] = A[M,1024] * Wt[1024,1024]^T (+bias)(+SiLU)
//    Block tile 128x128, 8 waves (4x2), wave tile 32x64, K-step 32.
//    Double-buffered LDS fed by GLOBAL_LOAD_ASYNC_TO_LDS_B128.
// ---------------------------------------------------------------------------
template <int ACT, bool HASB, bool WF, bool WB>
__global__ __launch_bounds__(256) void tg_gemm(const unsigned short* __restrict__ A,
                                               const unsigned short* __restrict__ Wt,
                                               const float* __restrict__ bias,
                                               float* __restrict__ outF,
                                               unsigned short* __restrict__ outB) {
    constexpr int K = 1024, N = 1024;
    __shared__ alignas(16) unsigned short sA[2][128 * 40];
    __shared__ alignas(16) unsigned short sB[2][128 * 40];

    const int tid = threadIdx.x;
    const int wave = tid >> 5, lane = tid & 31;
    const int laneM = lane & 15, hi = (lane >> 4) & 1;
    const int waveM = wave & 3, waveN = wave >> 2;
    const int m0 = blockIdx.y * 128, n0 = blockIdx.x * 128;

    // this thread's two 16B chunks per tile: rows rowF and rowF+64, col chunk ccF
    const int rowF = tid >> 2, ccF = tid & 3;
    const unsigned short* Aptr = A + (size_t)(m0 + rowF) * K + ccF * 8;
    const unsigned short* Bptr = Wt + (size_t)(n0 + rowF) * K + ccF * 8;
    const unsigned laOff0 = (unsigned)(unsigned long long)&sA[0][rowF * 40 + ccF * 8];
    const unsigned lbOff0 = (unsigned)(unsigned long long)&sB[0][rowF * 40 + ccF * 8];
    const unsigned laOff1 = (unsigned)(unsigned long long)&sA[1][rowF * 40 + ccF * 8];
    const unsigned lbOff1 = (unsigned)(unsigned long long)&sB[1][rowF * 40 + ccF * 8];

    auto issueTile = [&](int buf, int k0) {
        unsigned la = buf ? laOff1 : laOff0;
        unsigned lb = buf ? lbOff1 : lbOff0;
        tg_async16(la,         Aptr + k0);
        tg_async16(la + 5120u, Aptr + 64 * K + k0);   // +64 rows * 80B
        tg_async16(lb,         Bptr + k0);
        tg_async16(lb + 5120u, Bptr + 64 * K + k0);
    };

    v8f acc[2][4];
    for (int i = 0; i < 2; ++i)
        for (int j = 0; j < 4; ++j)
            for (int e = 0; e < 8; ++e) acc[i][j][e] = 0.0f;

    issueTile(0, 0);                                   // prologue: tile 0 in flight

    for (int ks = 0; ks < 32; ++ks) {
        tg_wait_async0();                              // this wave's async done
        __syncthreads();                               // all waves' tile landed
        if (ks < 31) issueTile((ks + 1) & 1, (ks + 1) * 32);

        const unsigned short* cA = &sA[ks & 1][0];
        const unsigned short* cB = &sB[ks & 1][0];

        BfFrag fa[2], fb[4];
        for (int mi = 0; mi < 2; ++mi) {
            int off = (waveM * 32 + mi * 16 + laneM) * 40 + hi * 8;
            fa[mi].h[0] = *reinterpret_cast<const v8bf*>(&cA[off]);
            fa[mi].h[1] = *reinterpret_cast<const v8bf*>(&cA[off + 16]);
        }
        for (int ni = 0; ni < 4; ++ni) {
            int off = (waveN * 64 + ni * 16 + laneM) * 40 + hi * 8;
            fb[ni].h[0] = *reinterpret_cast<const v8bf*>(&cB[off]);
            fb[ni].h[1] = *reinterpret_cast<const v8bf*>(&cB[off + 16]);
        }
        for (int mi = 0; mi < 2; ++mi)
            for (int ni = 0; ni < 4; ++ni)
                acc[mi][ni] = __builtin_amdgcn_wmma_f32_16x16x32_bf16(
                    false, fa[mi].v, false, fb[ni].v, (short)0, acc[mi][ni], false, false);
    }

    // epilogue: bias + SiLU, write f32 and/or bf16
    for (int mi = 0; mi < 2; ++mi) {
        int rbase = m0 + waveM * 32 + mi * 16 + hi * 8;
        for (int ni = 0; ni < 4; ++ni) {
            int c = n0 + waveN * 64 + ni * 16 + laneM;
            float bv = HASB ? bias[c] : 0.0f;
            for (int j = 0; j < 8; ++j) {
                float v = acc[mi][ni][j] + bv;
                if (ACT == 0) v = v * (1.0f / (1.0f + expf(-v)));   // SiLU
                size_t o = (size_t)(rbase + j) * N + c;
                if (WF) outF[o] = v;
                if (WB) outB[o] = f2bf(v);
            }
        }
    }
}

// ---------------------------------------------------------------------------
// 4) he = logmap0(expmap0(x)) per row (radial scaling), f32 in -> bf16 out
// ---------------------------------------------------------------------------
__global__ __launch_bounds__(256) void tg_he(const float* __restrict__ x,
                                             unsigned short* __restrict__ he) {
    __shared__ float red[256];
    __shared__ float sscale;
    int row = blockIdx.x, tid = threadIdx.x;
    float4 v = reinterpret_cast<const float4*>(x)[(size_t)row * 256 + tid];
    red[tid] = v.x * v.x + v.y * v.y + v.z * v.z + v.w * v.w;
    __syncthreads();
    for (int s = 128; s > 0; s >>= 1) {
        if (tid < s) red[tid] += red[tid + s];
        __syncthreads();
    }
    if (tid == 0) {
        float n1 = fmaxf(sqrtf(red[0]), EPSF);
        float t = tanhf(n1);
        float n2 = fmaxf(t, EPSF);
        sscale = (atanhf(fminf(n2, MAXNF)) / n2) * (t / n1);
    }
    __syncthreads();
    float sc = sscale;
    ushort4 o;
    o.x = f2bf(v.x * sc); o.y = f2bf(v.y * sc);
    o.z = f2bf(v.z * sc); o.w = f2bf(v.w * sc);
    reinterpret_cast<ushort4*>(he)[(size_t)row * 256 + tid] = o;
}

// ---------------------------------------------------------------------------
// 5) agg[b,c,d] = sum_r adj[b,r,c] * m[b,r,d] + gcn_b3[d]  via WMMA.
//    One block per (batch, 128-wide d slice). M=77->80, K=77->96 in LDS.
// ---------------------------------------------------------------------------
__global__ __launch_bounds__(256) void tg_agg(const int* __restrict__ edge,
                                              const unsigned short* __restrict__ mB,
                                              const float* __restrict__ bias3,
                                              float* __restrict__ aggF) {
    __shared__ alignas(16) unsigned short sAdj[80 * 104];   // adj^T tile [c][r]
    __shared__ alignas(16) unsigned short sM[128 * 104];    // m^T  tile [n][r]

    const int tid = threadIdx.x;
    const int wave = tid >> 5, lane = tid & 31;
    const int laneM = lane & 15, hi = (lane >> 4) & 1;
    const int b = blockIdx.y;
    const int n0 = blockIdx.x * 128;

    for (int idx = tid; idx < 80 * 96; idx += 256) {      // 30 exact iters
        int c = idx / 96, r = idx - c * 96;
        unsigned short v = 0;
        if (c < TG_S && r < TG_S)
            v = (edge[((size_t)b * TG_S + r) * TG_S + c] != 0) ? (unsigned short)0x3F80 : 0;
        sAdj[c * 104 + r] = v;
    }
    for (int idx = tid; idx < 96 * 128; idx += 256) {     // 48 exact iters
        int r = idx >> 7, n = idx & 127;
        unsigned short v = 0;
        if (r < TG_S) v = mB[((size_t)b * TG_S + r) * TG_D + n0 + n];
        sM[n * 104 + r] = v;
    }
    __syncthreads();

    v8f acc[5];
    for (int i = 0; i < 5; ++i)
        for (int e = 0; e < 8; ++e) acc[i][e] = 0.0f;

    for (int kk = 0; kk < 3; ++kk) {
        int k = kk * 32;
        BfFrag fb;
        int offB = (wave * 16 + laneM) * 104 + k + hi * 8;
        fb.h[0] = *reinterpret_cast<const v8bf*>(&sM[offB]);
        fb.h[1] = *reinterpret_cast<const v8bf*>(&sM[offB + 16]);
        for (int mi = 0; mi < 5; ++mi) {
            BfFrag fa;
            int offA = (mi * 16 + laneM) * 104 + k + hi * 8;
            fa.h[0] = *reinterpret_cast<const v8bf*>(&sAdj[offA]);
            fa.h[1] = *reinterpret_cast<const v8bf*>(&sAdj[offA + 16]);
            acc[mi] = __builtin_amdgcn_wmma_f32_16x16x32_bf16(
                false, fa.v, false, fb.v, (short)0, acc[mi], false, false);
        }
    }

    int d = n0 + wave * 16 + laneM;
    float bv = bias3[d];
    for (int mi = 0; mi < 5; ++mi) {
        for (int j = 0; j < 8; ++j) {
            int c = mi * 16 + j + hi * 8;
            if (c < TG_S)
                aggF[((size_t)b * TG_S + c) * TG_D + d] = acc[mi][j] + bv;
        }
    }
}

// ---------------------------------------------------------------------------
// 6) out = logmap0(expmap0(leaky(logmap0(expmap0(agg)))))  per row.
// ---------------------------------------------------------------------------
__global__ __launch_bounds__(256) void tg_final(const float* __restrict__ agg,
                                                float* __restrict__ out) {
    __shared__ float red[256];
    __shared__ float sc12s, sc34s;
    int row = blockIdx.x, tid = threadIdx.x;
    float4 v = reinterpret_cast<const float4*>(agg)[(size_t)row * 256 + tid];
    red[tid] = v.x * v.x + v.y * v.y + v.z * v.z + v.w * v.w;
    __syncthreads();
    for (int s = 128; s > 0; s >>= 1) {
        if (tid < s) red[tid] += red[tid + s];
        __syncthreads();
    }
    if (tid == 0) {
        float n1 = fmaxf(sqrtf(red[0]), EPSF);
        float t1 = tanhf(n1);
        float n2 = fmaxf(t1, EPSF);
        sc12s = (t1 / n1) * (atanhf(fminf(n2, MAXNF)) / n2);   // expmap0 then logmap0
    }
    __syncthreads();
    float sc12 = sc12s;
    float4 u;
    u.x = v.x * sc12; u.y = v.y * sc12; u.z = v.z * sc12; u.w = v.w * sc12;
    u.x = u.x > 0.0f ? u.x : 0.01f * u.x;
    u.y = u.y > 0.0f ? u.y : 0.01f * u.y;
    u.z = u.z > 0.0f ? u.z : 0.01f * u.z;
    u.w = u.w > 0.0f ? u.w : 0.01f * u.w;
    __syncthreads();
    red[tid] = u.x * u.x + u.y * u.y + u.z * u.z + u.w * u.w;
    __syncthreads();
    for (int s = 128; s > 0; s >>= 1) {
        if (tid < s) red[tid] += red[tid + s];
        __syncthreads();
    }
    if (tid == 0) {
        float n3 = fmaxf(sqrtf(red[0]), EPSF);
        float t3 = tanhf(n3);
        float n4 = fmaxf(t3, EPSF);
        sc34s = (t3 / n3) * (atanhf(fminf(n4, MAXNF)) / n4);   // expmap0 then logmap0
    }
    __syncthreads();
    float sc34 = sc34s;
    float4 o;
    o.x = u.x * sc34; o.y = u.y * sc34; o.z = u.z * sc34; o.w = u.w * sc34;
    reinterpret_cast<float4*>(out)[(size_t)row * 256 + tid] = o;
}

// ---------------------------------------------------------------------------
// Host launcher
// ---------------------------------------------------------------------------
extern "C" void kernel_launch(void* const* d_in, const int* in_sizes, int n_in,
                              void* d_out, int out_size, void* d_ws, size_t ws_size,
                              hipStream_t stream) {
    (void)in_sizes; (void)n_in; (void)out_size; (void)ws_size;

    const long long* tokens = (const long long*)d_in[0];   // int64 per reference
    const int*       edge   = (const int*)d_in[1];
    const float*     tok_w  = (const float*)d_in[2];
    const float*     pos_w  = (const float*)d_in[3];
    const float*     mlp_w  = (const float*)d_in[4];
    const float*     mlp_b  = (const float*)d_in[5];
    const float*     gcn_w  = (const float*)d_in[6];
    const float*     gcn_b  = (const float*)d_in[7];

    char* ws = (char*)d_ws;
    const size_t wtBytes  = (size_t)6 * 1024 * 1024 * 2;       // 12,582,912
    const size_t actBytes = (size_t)TG_M * TG_D * 2;           // 40,370,176
    unsigned short* wt   = (unsigned short*)ws;
    unsigned short* actA = (unsigned short*)(ws + wtBytes);
    unsigned short* actB = (unsigned short*)(ws + wtBytes + actBytes);
    float*          xbuf = (float*)(ws + wtBytes + 2 * actBytes);  // 80.7 MB f32

    // 1) weights -> bf16, transposed
    tg_convert_w<<<24576, 256, 0, stream>>>(mlp_w, gcn_w, wt);
    // 2) embedding -> bf16
    tg_embed<<<TG_M * TG_D / 256, 256, 0, stream>>>(tokens, tok_w, pos_w, actA);

    dim3 ggrid(TG_D / 128, TG_M / 128);   // 8 x 154
    unsigned short* aPing = actA;
    unsigned short* aPong = actB;
    // 3) MLP layers 0..3: GEMM + bias + SiLU -> bf16
    for (int i = 0; i < 4; ++i) {
        tg_gemm<0, true, false, true><<<ggrid, 256, 0, stream>>>(
            aPing, wt + (size_t)i * 1048576, mlp_b + (size_t)i * TG_D, nullptr, aPong);
        unsigned short* t = aPing; aPing = aPong; aPong = t;
    }
    // layer 4: GEMM + bias, no act -> f32
    tg_gemm<1, true, true, false><<<ggrid, 256, 0, stream>>>(
        aPing, wt + (size_t)4 * 1048576, mlp_b + (size_t)4 * TG_D, xbuf, nullptr);
    // 4) he = logmap0(expmap0(x)) -> bf16 (into aPong)
    tg_he<<<TG_M, 256, 0, stream>>>(xbuf, aPong);
    // 5) m = he @ gcn_w[3] (no bias, no act) -> bf16 (into aPing)
    tg_gemm<1, false, false, true><<<ggrid, 256, 0, stream>>>(
        aPong, wt + (size_t)5 * 1048576, nullptr, nullptr, aPing);
    // 6) agg = adj^T @ m + gcn_b[3] -> f32 (reuse xbuf)
    dim3 agrid(TG_D / 128, TG_B);         // 8 x 256
    tg_agg<<<agrid, 256, 0, stream>>>(edge, aPing, gcn_b + 3 * TG_D, xbuf);
    // 7) final radial chain -> out
    tg_final<<<TG_M, 256, 0, stream>>>(xbuf, (float*)d_out);
}
